// APLoss_12369505812832
// MI455X (gfx1250) — compile-verified, matrix operand order checked by reading, not run on hardware
//
#include <hip/hip_runtime.h>
#include <hip/hip_bf16.h>

// ---------------------------------------------------------------------------
// AP-loss forward for MI455X (gfx1250, wave32).
//
// Roofline: 16 MB traffic @ 23.3 TB/s => ~0.7us floor. Brute force is
// N*FG = 512M clamp-FMAs (VALU bound, ~10-30x over floor). We instead use the
// piecewise-linear structure of clamp((l-f+1)/2,0,1) over the *sorted*
// thresholds: per element, one "+1 below lo" point update + one banded linear
// ramp expressed as difference-array range updates => O(N log FG).
// The fg-vs-fg term (a_i) is a true dense 256x256 clamped matrix row-sum and
// is reduced with V_WMMA_F32_16X16X32_F16 (ones-matrix A => column sums).
// ---------------------------------------------------------------------------

typedef __attribute__((ext_vector_type(16))) _Float16 v16h;
typedef __attribute__((ext_vector_type(8)))  float    v8f;

#define FG 256

// Workspace layout (float element offsets inside d_ws)
#define WS_CNT_OFF   0      // [1]   int append counter
#define WS_RAW_OFF   64     // [256] fg logits, unsorted
#define WS_SORT_OFF  320    // [256] fg logits, ascending
#define WS_CNTARR    576    // [257] point counts at 'lo'
#define WS_VARR      840    // [257] diff array for (l+1)/2 range adds
#define WS_WARR      1104   // [257] diff array for 1/2 range adds
#define WS_TOTAL     1408

// ---------------- init: zero all accumulators (ws is poisoned 0xAA) --------
__global__ void ap_init(float* ws) {
    int t = threadIdx.x;
    if (t == 0) *((int*)(ws + WS_CNT_OFF)) = 0;
    for (int i = t; i < 257; i += blockDim.x) {
        ws[WS_CNTARR + i] = 0.0f;
        ws[WS_VARR   + i] = 0.0f;
        ws[WS_WARR   + i] = 0.0f;
    }
}

// ---------------- collect the FG foreground logits --------------------------
__global__ void ap_collect(const float* __restrict__ logits,
                           const int* __restrict__ targets,
                           float* ws, int n) {
    int* cnt = (int*)(ws + WS_CNT_OFF);
    float* raw = ws + WS_RAW_OFF;
    int stride = gridDim.x * blockDim.x;
    for (int i = blockIdx.x * blockDim.x + threadIdx.x; i < n; i += stride) {
        if (targets[i] == 1) {
            int p = atomicAdd(cnt, 1);
            if (p < FG) raw[p] = logits[i];
        }
    }
}

// ---------------- bitonic sort of the 256 fg logits (1 workgroup) ----------
__global__ void ap_sort(float* ws) {
    __shared__ float s[FG];
    int tid = threadIdx.x;
    s[tid] = ws[WS_RAW_OFF + tid];
    __syncthreads();
    for (int k = 2; k <= FG; k <<= 1) {
        for (int j = k >> 1; j > 0; j >>= 1) {
            int ixj = tid ^ j;
            if (ixj > tid) {
                bool up = ((tid & k) == 0);
                float a = s[tid], b = s[ixj];
                if ((a > b) == up) { s[tid] = b; s[ixj] = a; }
            }
            __syncthreads();
        }
    }
    ws[WS_SORT_OFF + tid] = s[tid];
}

// first idx with sf[idx] > x   (count of elements <= x)
__device__ __forceinline__ int ap_ub(const float* sf, float x) {
    int lo = 0, hi = FG;
    while (lo < hi) { int m = (lo + hi) >> 1; if (sf[m] <= x) lo = m + 1; else hi = m; }
    return lo;
}
// first idx with sf[idx] >= x  (count of elements < x)
__device__ __forceinline__ int ap_lb(const float* sf, float x) {
    int lo = 0, hi = FG;
    while (lo < hi) { int m = (lo + hi) >> 1; if (sf[m] < x) lo = m + 1; else hi = m; }
    return lo;
}

// ---------------- heavy phase: background difference-array accumulation ----
__global__ void ap_bg(const float* __restrict__ logits,
                      const int* __restrict__ targets,
                      float* ws, int n) {
    __shared__ float s_f[FG];
    __shared__ float s_cnt[257];
    __shared__ float s_V[257];
    __shared__ float s_W[257];

    int tid = threadIdx.x;
    for (int i = tid; i < 257; i += blockDim.x) {
        s_cnt[i] = 0.0f; s_V[i] = 0.0f; s_W[i] = 0.0f;
    }
    if (tid < FG) s_f[tid] = ws[WS_SORT_OFF + tid];
    __syncthreads();

    int gtid   = blockIdx.x * blockDim.x + tid;
    int stride = gridDim.x * blockDim.x;

    auto process = [&](float l, int tgt) {
        if (tgt != 0) return;                 // fg entries are excluded from b
        int lo = ap_ub(s_f, l - 1.0f);        // thresholds <= l-1 get +1
        int hi = ap_lb(s_f, l + 1.0f);        // band: lo..hi-1 gets ramp
        atomicAdd(&s_cnt[lo], 1.0f);
        if (hi > lo) {
            float v = (l + 1.0f) * 0.5f;      // (l+1-f_i)/2 = v - f_i*0.5
            atomicAdd(&s_V[lo],  v);  atomicAdd(&s_V[hi], -v);
            atomicAdd(&s_W[lo],  0.5f); atomicAdd(&s_W[hi], -0.5f);
        }
    };

    int n4 = n >> 2;
    const float4* l4 = (const float4*)logits;
    const int4*   t4 = (const int4*)targets;
    for (int i = gtid; i < n4; i += stride) {
        float4 lv = l4[i];
        int4   tv = t4[i];
        process(lv.x, tv.x); process(lv.y, tv.y);
        process(lv.z, tv.z); process(lv.w, tv.w);
    }
    for (int i = (n4 << 2) + gtid; i < n; i += stride)
        process(logits[i], targets[i]);

    __syncthreads();
    for (int i = tid; i < 257; i += blockDim.x) {
        if (s_cnt[i] != 0.0f) atomicAdd(&ws[WS_CNTARR + i], s_cnt[i]);
        if (s_V[i]   != 0.0f) atomicAdd(&ws[WS_VARR   + i], s_V[i]);
        if (s_W[i]   != 0.0f) atomicAdd(&ws[WS_WARR   + i], s_W[i]);
    }
}

// ---------------- finalize: prefix sums + WMMA a_i + precision scan --------
__global__ void ap_finalize(float* ws, float* out) {
    __shared__ float s_f[FG];
    __shared__ float s_cnt[257];
    __shared__ float s_V[257];
    __shared__ float s_W[257];
    __shared__ float s_a[FG];
    __shared__ float s_b[FG];

    int tid = threadIdx.x;
    if (tid < FG) s_f[tid] = ws[WS_SORT_OFF + tid];
    for (int i = tid; i < 257; i += blockDim.x) {
        s_cnt[i] = ws[WS_CNTARR + i];
        s_V[i]   = ws[WS_VARR   + i];
        s_W[i]   = ws[WS_WARR   + i];
    }
    __syncthreads();

    // b_i = (#bg fully above band) + prefix(V)_i - f_i * prefix(W)_i
    if (tid == 0) {
        float run = 0.0f;
        for (int j = FG; j >= 1; --j) {       // suffix sum of point counts
            run += s_cnt[j];
            s_b[j - 1] = run;
        }
        float pv = 0.0f, pw = 0.0f;
        for (int j = 0; j < FG; ++j) {        // inclusive prefix of diff arrays
            pv += s_V[j];
            pw += s_W[j];
            s_b[j] += pv - s_f[j] * pw;
        }
    }

    // a_i via WMMA: B[k,n] = clamp((f[j0+k]-f[i0+n])*0.5+0.5), A = ones
    // => D[m,n] = column sums. 8 waves x 2 column-tiles x 8 K-chunks.
    {
        int lane  = tid & 31;
        int wave  = tid >> 5;
        int col   = lane & 15;           // B/D column owned by this lane
        int khalf = (lane >> 4) * 16;    // lanes 16-31 hold K=16..31 of chunk

        v16h ones;
        #pragma unroll
        for (int v = 0; v < 16; ++v) ones[v] = (_Float16)1.0f;

        #pragma unroll
        for (int t = 0; t < 2; ++t) {
            int i0  = (wave * 2 + t) * 16;
            float thr = s_f[i0 + col];
            v8f acc = {};
            #pragma unroll
            for (int kc = 0; kc < 8; ++kc) {
                int j0 = kc * 32 + khalf;
                v16h bmat;
                #pragma unroll
                for (int v = 0; v < 8; ++v) {
                    float x0 = (s_f[j0 + 2 * v]     - thr) * 0.5f + 0.5f;
                    float x1 = (s_f[j0 + 2 * v + 1] - thr) * 0.5f + 0.5f;
                    x0 = fminf(fmaxf(x0, 0.0f), 1.0f);
                    x1 = fminf(fmaxf(x1, 0.0f), 1.0f);
                    bmat[2 * v]     = (_Float16)x0;
                    bmat[2 * v + 1] = (_Float16)x1;
                }
                acc = __builtin_amdgcn_wmma_f32_16x16x32_f16(
                    false, ones, false, bmat, (short)0, acc, false, false);
            }
            if (lane < 16) s_a[i0 + col] = 0.5f + acc[0];
        }
    }
    __syncthreads();

    if (tid == 0) {
        float maxp = 0.0f, sum = 0.0f;
        for (int i = 0; i < FG; ++i) {
            float a = s_a[i];
            float cur = a / (a + s_b[i]);
            maxp = fmaxf(maxp, cur);
            sum += maxp;
        }
        out[0] = 1.0f - sum / (float)FG;
    }
}

extern "C" void kernel_launch(void* const* d_in, const int* in_sizes, int n_in,
                              void* d_out, int out_size, void* d_ws, size_t ws_size,
                              hipStream_t stream) {
    const float* logits  = (const float*)d_in[0];
    const int*   targets = (const int*)d_in[1];
    float* ws  = (float*)d_ws;
    float* out = (float*)d_out;
    int n = in_sizes[0];

    ap_init<<<1, 1024, 0, stream>>>(ws);
    ap_collect<<<2048, 256, 0, stream>>>(logits, targets, ws, n);
    ap_sort<<<1, FG, 0, stream>>>(ws);
    ap_bg<<<1280, 256, 0, stream>>>(logits, targets, ws, n);
    ap_finalize<<<1, 256, 0, stream>>>(ws, out);
}
